// CrossDimensionAttention_3985729651637
// MI455X (gfx1250) — compile-verified, matrix operand order checked by reading, not run on hardware
//
#include <hip/hip_runtime.h>
#include <hip/hip_bf16.h>

// ---------------------------------------------------------------------------
// CrossDimensionAttention fused kernel for gfx1250 (MI455X), wave32 + WMMA.
//   B=8, T=96, L=128, D=256, H=2, HD=128, FACTOR=10, WINDOW=5, DFF=1024
// One block per n=(b,l). 256 threads = 8 wave32. All GEMMs via
// v_wmma_f32_16x16x32_f16 (f32 accumulate); LN / softmax / gelu in f32 VALU.
// x tile staged LDS-side by the Tensor Data Mover (tensor_load_to_lds).
// ---------------------------------------------------------------------------

typedef _Float16 h16;
typedef __attribute__((ext_vector_type(16))) _Float16 v16h;
typedef __attribute__((ext_vector_type(8)))  _Float16 v8h;
typedef __attribute__((ext_vector_type(8)))  float    v8f;
typedef __attribute__((ext_vector_type(4)))  float    v4f;
typedef __attribute__((ext_vector_type(4)))  unsigned int v4u;
typedef __attribute__((ext_vector_type(8)))  int      v8i;
typedef __attribute__((ext_vector_type(4)))  int      v4i;

#define CD_B      8
#define CD_T      96
#define CD_L      128
#define CD_D      256
#define CD_H      2
#define CD_HD     128
#define CD_F      10
#define CD_WIN    5
#define CD_DFF    1024
#define LDX       272          // halves per row of [96][256] f16 buffers (16B-aligned rows, padded)
#define LDH       136          // halves per row of H-chunk [96][128]
#define LDACC     260          // floats per row of f32 accum buffer
#define CD_SCALE  0.0883883476483184f   // HD^-0.5
#define CD_EPS    1e-5f

// ---- shared-memory arena layout (bytes) -----------------------------------
// sXp   : [96][LDX] f16                        @ 0        (52224)
// arena : @ 52224
//   phase 0   : sStage f32 [96][256] @ +0 (98304)   (TDM destination)
//   phase MLP : sAcc f32 [96][LDACC] @ +0 (99840) ; sH [96][LDH] f16 @ +99840 (26112)
//   phase QKV : sK @ +0 (52224) ; sV @ +52224 ; sQ @ +104448 ;
//               sR [16][LDX] @ +156672 (8704) ; sSend f32 [2][16][128] @ +165376 (16384)
//   sRecv [96][LDX] f16 @ +181760 (52224)
// total = 286208 bytes  (< 320KB WGP LDS)
#define SMEM_BYTES 286208
#define ARENA_OFF  52224u

__device__ __forceinline__ v8f zero8() {
  v8f z;
#pragma unroll
  for (int i = 0; i < 8; ++i) z[i] = 0.0f;
  return z;
}

__device__ __forceinline__ v8f wmma_f16(v16h a, v16h b, v8f c) {
  // D = A(16x32) * B(32x16) + C ; f32 accumulate
  return __builtin_amdgcn_wmma_f32_16x16x32_f16(false, a, false, b, (short)0, c,
                                                false, false);
}

// Load a 16x32 f16 fragment (A-style) from a row-major matrix with leading
// dimension `ld` (in halves). Per ISA 7.12.2: lane holds row = lane%16;
// lanes 0-15 hold K 0-7 & 16-23, lanes 16-31 hold K 8-15 & 24-31.
// The same loader serves B fragments when given rows of B^T (i.e. W[out,in]).
__device__ __forceinline__ v16h load_frag(const h16* p0, int ld, int lane) {
  const h16* p = p0 + (lane & 15) * ld + ((lane >> 4) << 3);
  v8h lo = *(const v8h*)(p);
  v8h hi = *(const v8h*)(p + 16);
  v16h r;
#pragma unroll
  for (int i = 0; i < 8; ++i) { r[i] = lo[i]; r[i + 8] = hi[i]; }
  return r;
}

__device__ __forceinline__ float wave_max(float v) {
#pragma unroll
  for (int off = 16; off > 0; off >>= 1) v = fmaxf(v, __shfl_xor(v, off, 32));
  return v;
}
__device__ __forceinline__ float wave_sum(float v) {
#pragma unroll
  for (int off = 16; off > 0; off >>= 1) v += __shfl_xor(v, off, 32);
  return v;
}

__device__ __forceinline__ float gelu_f(float v) {
  return 0.5f * v * (1.0f + erff(v * 0.70710678118654752f));
}

// ---------------------------------------------------------------------------
// Tensor Data Mover: stage the [96 rows x 256 f32] x-tile (row stride L*D)
// from global into contiguous LDS at byte offset `lds_off`.
// D# per CDNA5 ISA (2D, data_size=4B).
// Must be issued from a single wave (EXEC is ignored by TDM).
// ---------------------------------------------------------------------------
__device__ __forceinline__ void tdm_load_x_tile(const float* gsrc,
                                                unsigned int lds_off) {
  unsigned long long ga = (unsigned long long)(uintptr_t)gsrc;
  v4u g0;
  g0.x = 1u;                                    // count=1, user descriptor
  g0.y = lds_off;                               // lds_addr (bytes)
  g0.z = (unsigned int)(ga & 0xffffffffull);    // global_addr[31:0]
  g0.w = (unsigned int)((ga >> 32) & 0x1ffffffull) | (2u << 30); // [56:32]+type=2
  v8i g1;
  g1[0] = (int)(2u << 16);      // data_size = 4 bytes; wg_mask=0; no flags
  g1[1] = (int)(256u << 16);    // tensor_dim0 = 256      (bits 79:48)
  g1[2] = (int)(96u << 16);     // tensor_dim1 = 96       (bits 111:80)
  g1[3] = (int)(256u << 16);    // tile_dim0   = 256      (bits 127:112)
  g1[4] = 96;                   // tile_dim1   = 96       (bits 143:128)
  g1[5] = (int)(CD_L * CD_D);   // tensor_dim0_stride = 32768 (bits 207:160)
  g1[6] = 0;
  g1[7] = 0;
  v4i gz;
  gz[0] = 0; gz[1] = 0; gz[2] = 0; gz[3] = 0;
#if defined(__clang_major__) && (__clang_major__ >= 23)
  v8i gz8;
#pragma unroll
  for (int i = 0; i < 8; ++i) gz8[i] = 0;
  __builtin_amdgcn_tensor_load_to_lds(g0, g1, gz, gz, gz8, 0);
#else
  __builtin_amdgcn_tensor_load_to_lds(g0, g1, gz, gz, 0);
#endif
}

// ---------------------------------------------------------------------------
// MLP: sAcc[96][LDACC] = gelu(sA @ W1^T + b1) @ W2^T + b2
// W1: [DFF][256] f16 row-major (global, L2-resident), W2: [256][DFF] f16.
// DFF processed in 8 chunks of 128; 2nd-GEMM accumulators live in registers:
// wave wv owns output column tiles {2wv, 2wv+1} for all 6 row tiles.
// ---------------------------------------------------------------------------
__device__ __forceinline__ void mlp_block(const h16* __restrict__ sA,
                                          const h16* __restrict__ W1,
                                          const float* __restrict__ b1,
                                          const h16* __restrict__ W2,
                                          const float* __restrict__ b2,
                                          h16* sH, float* sAcc, int wv, int lane) {
  v8f acc[6][2];
#pragma unroll
  for (int mt = 0; mt < 6; ++mt) { acc[mt][0] = zero8(); acc[mt][1] = zero8(); }
  const int n  = lane & 15;
  const int m0 = (lane >> 4) << 3;

  for (int c = 0; c < 8; ++c) {
    // ---- H chunk: wave wv computes column tile ht = wv of H [96][128] ----
#pragma unroll
    for (int mt = 0; mt < 6; ++mt) {
      v8f hc = zero8();
#pragma unroll
      for (int ks = 0; ks < 8; ++ks) {
        v16h a = load_frag(sA + mt * 16 * LDX + ks * 32, LDX, lane);
        v16h b = load_frag(W1 + (c * 128 + wv * 16) * 256 + ks * 32, 256, lane);
        hc = wmma_f16(a, b, hc);
      }
      float bb = b1[c * 128 + wv * 16 + n];
#pragma unroll
      for (int r = 0; r < 8; ++r) {
        float v = gelu_f(hc[r] + bb);
        sH[(mt * 16 + m0 + r) * LDH + wv * 16 + n] = (h16)v;
      }
    }
    __syncthreads();
    // ---- accumulate: out tiles (mt, 2wv+j) += H_chunk @ W2_chunk^T ----
#pragma unroll
    for (int j = 0; j < 2; ++j) {
      const int nt = wv * 2 + j;
#pragma unroll
      for (int mt = 0; mt < 6; ++mt) {
#pragma unroll
        for (int ks = 0; ks < 4; ++ks) {
          v16h a = load_frag(sH + mt * 16 * LDH + ks * 32, LDH, lane);
          v16h b = load_frag(W2 + nt * 16 * CD_DFF + c * 128 + ks * 32, CD_DFF, lane);
          acc[mt][j] = wmma_f16(a, b, acc[mt][j]);
        }
      }
    }
    __syncthreads();
  }
  // ---- spill accumulators (+bias) to f32 LDS ----
#pragma unroll
  for (int j = 0; j < 2; ++j) {
    int ng = (wv * 2 + j) * 16 + n;
    float bb = b2[ng];
#pragma unroll
    for (int mt = 0; mt < 6; ++mt)
#pragma unroll
      for (int r = 0; r < 8; ++r)
        sAcc[(mt * 16 + m0 + r) * LDACC + ng] = acc[mt][j][r] + bb;
  }
}

// sDst[96][LDX] = sA[96][LDX] @ W^T + bias, W: [256][256] f16 row-major.
__device__ __forceinline__ void proj256(const h16* __restrict__ sA,
                                        const h16* __restrict__ Wg,
                                        const float* __restrict__ bias,
                                        h16* sDst, int wv, int lane) {
  const int n = lane & 15, m0 = (lane >> 4) << 3;
#pragma unroll
  for (int j = 0; j < 2; ++j) {
    int nt = wv * 2 + j;
    float bb = bias[nt * 16 + n];
#pragma unroll
    for (int mt = 0; mt < 6; ++mt) {
      v8f c = zero8();
#pragma unroll
      for (int ks = 0; ks < 8; ++ks) {
        v16h a = load_frag(sA + mt * 16 * LDX + ks * 32, LDX, lane);
        v16h b = load_frag(Wg + nt * 16 * 256 + ks * 32, 256, lane);
        c = wmma_f16(a, b, c);
      }
#pragma unroll
      for (int r = 0; r < 8; ++r)
        sDst[(mt * 16 + m0 + r) * LDX + nt * 16 + n] = (h16)(c[r] + bb);
    }
  }
}

// Router projection: sR[16][LDX] = rowsA(16 rows, ld 256) @ W^T + bias
__device__ __forceinline__ void projR(const h16* __restrict__ sAsrc,
                                      const h16* __restrict__ Wg,
                                      const float* __restrict__ bias,
                                      h16* sR, int wv, int lane) {
  const int n = lane & 15, m0 = (lane >> 4) << 3;
#pragma unroll
  for (int j = 0; j < 2; ++j) {
    int nt = wv * 2 + j;
    float bb = bias[nt * 16 + n];
    v8f c = zero8();
#pragma unroll
    for (int ks = 0; ks < 8; ++ks) {
      v16h a = load_frag(sAsrc + ks * 32, 256, lane);
      v16h b = load_frag(Wg + nt * 16 * 256 + ks * 32, 256, lane);
      c = wmma_f16(a, b, c);
    }
#pragma unroll
    for (int r = 0; r < 8; ++r)
      sR[(m0 + r) * LDX + nt * 16 + n] = (h16)(c[r] + bb);
  }
}

// ---------------------------------------------------------------------------
__global__ __launch_bounds__(256)
void cda_main(const float* __restrict__ x, const h16* __restrict__ rt16,
              const h16* __restrict__ wr16, const float* __restrict__ br,
              const h16* __restrict__ wk16, const float* __restrict__ bk,
              const h16* __restrict__ wv16, const float* __restrict__ bv,
              const h16* __restrict__ wq16, const float* __restrict__ bq,
              const float* __restrict__ g1, const float* __restrict__ be1,
              const float* __restrict__ g2, const float* __restrict__ be2,
              const float* __restrict__ g3, const float* __restrict__ be3,
              const h16* __restrict__ m1w1, const float* __restrict__ m1b1,
              const h16* __restrict__ m1w2, const float* __restrict__ m1b2,
              const h16* __restrict__ m2w1, const float* __restrict__ m2b1,
              const h16* __restrict__ m2w2, const float* __restrict__ m2b2,
              float* __restrict__ out) {
  extern __shared__ char smem[];
  h16*   sXp   = (h16*)(smem);                    // [96][LDX]
  char*  arena = smem + ARENA_OFF;
  float* sStage= (float*)(arena);                 // [96][256] TDM staging
  float* sAcc  = (float*)(arena);                 // [96][LDACC]
  h16*   sH    = (h16*)(arena + 99840);           // [96][LDH]
  h16*   sK    = (h16*)(arena);                   // [96][LDX]
  h16*   sV    = (h16*)(arena + 52224);           // [96][LDX]
  h16*   sQ    = (h16*)(arena + 104448);          // [96][LDX]
  h16*   sR    = (h16*)(arena + 156672);          // [16][LDX]
  float* sSend = (float*)(arena + 165376);        // [2][16][128] f32
  h16*   sRecv = (h16*)(arena + 181760);          // [96][LDX]

  const int tid  = threadIdx.x;
  const int wv   = tid >> 5;
  const int lane = tid & 31;
  const int n    = blockIdx.x;           // n = b*L + l
  const int b    = n >> 7;
  const int l    = n & (CD_L - 1);
  const float* xg = x + ((size_t)b * CD_T * CD_L + (size_t)l) * CD_D;

  __builtin_prefetch(m1w1, 0, 3);
  __builtin_prefetch(m1w2, 0, 3);

  // ---- phase 0: TDM-stage x tile (f32) into LDS, wave 0 issues DMA ----
  if (wv == 0) {
    tdm_load_x_tile(xg, ARENA_OFF);
    __builtin_amdgcn_s_wait_tensorcnt(0);
  }
  __syncthreads();

  // ---- phase 1: convert staged x -> f16 sXp ---------------------------
  for (int idx = tid; idx < CD_T * CD_D; idx += 256) {
    int r = idx >> 8, d = idx & 255;
    sXp[r * LDX + d] = (h16)sStage[idx];
  }
  __syncthreads();

  // ---- phase 2: MLP1 (overwrites staging region with sAcc) ------------
  mlp_block(sXp, m1w1, m1b1, m1w2, m1b2, sH, sAcc, wv, lane);
  __syncthreads();

  // ---- phase 3: x_proj = LN1(x + MLP1(x)) -> sXp (f16) ----------------
  for (int r = wv; r < CD_T; r += 8) {
    const float* xr = xg + (size_t)r * CD_L * CD_D;
    float vals[8], s0 = 0.f, s1 = 0.f;
#pragma unroll
    for (int q = 0; q < 8; ++q) {
      int d = lane + q * 32;
      float v = xr[d] + sAcc[r * LDACC + d];
      vals[q] = v; s0 += v; s1 += v * v;
    }
    s0 = wave_sum(s0); s1 = wave_sum(s1);
    float mean = s0 * (1.0f / 256.0f);
    float inv  = rsqrtf(s1 * (1.0f / 256.0f) - mean * mean + CD_EPS);
#pragma unroll
    for (int q = 0; q < 8; ++q) {
      int d = lane + q * 32;
      sXp[r * LDX + d] = (h16)((vals[q] - mean) * inv * g1[d] + be1[d]);
    }
  }
  __syncthreads();

  // ---- phase 4: K / V / Q / router projections ------------------------
  proj256(sXp, wk16, bk, sK, wv, lane);
  proj256(sXp, wv16, bv, sV, wv, lane);
  proj256(sXp, wq16, bq, sQ, wv, lane);
  projR(rt16 + (size_t)l * 16 * 256, wr16, br, sR, wv, lane);
  __syncthreads();

  // ---- phase 5: sender = sparse_attn(router_proj, K, V) ---------------
  for (int task = wv; task < CD_H * CD_F; task += 8) {
    int h = task / CD_F, i = task % CD_F;
    int jlo = (i > CD_WIN) ? i - CD_WIN : 0;
    int jhi = i + CD_WIN;                 // i<10 -> always < 96
    int cnt = jhi - jlo + 1;
    float s = -1e30f;
    if (lane < cnt) {
      const v8h* rp = (const v8h*)(sR + i * LDX + h * CD_HD);
      const v8h* kp = (const v8h*)(sK + (jlo + lane) * LDX + h * CD_HD);
      float a = 0.f;
#pragma unroll
      for (int d8 = 0; d8 < CD_HD / 8; ++d8) {
        v8h rv = rp[d8], kv = kp[d8];
#pragma unroll
        for (int u = 0; u < 8; ++u) a += (float)rv[u] * (float)kv[u];
      }
      a *= CD_SCALE;
      s = fminf(fmaxf(a, -10000.f), 10000.f);
    }
    float m = wave_max(s);
    float e = (lane < cnt) ? __expf(s - m) : 0.f;
    float den = wave_sum(e);
    float w = e / den;
#pragma unroll
    for (int q = 0; q < 4; ++q) {
      int d = q * 32 + lane;
      float o = 0.f;
      for (int jj = 0; jj < cnt; ++jj)
        o += __shfl(w, jj, 32) * (float)sV[(jlo + jj) * LDX + h * CD_HD + d];
      sSend[(h * 16 + i) * CD_HD + d] = o;
    }
  }
  __syncthreads();

  // ---- phase 6: recv = sparse_attn(Q, sender, sender) -> sRecv (f16) --
  for (int task = wv; task < CD_H * CD_T; task += 8) {
    int h = task / CD_T, i = task % CD_T;
    int jlo, jhi;
    if (i >= CD_F) { jlo = 0; jhi = CD_F - 1; }
    else {
      jlo = (i > CD_WIN) ? i - CD_WIN : 0;
      jhi = i + CD_WIN; if (jhi > CD_F - 1) jhi = CD_F - 1;
    }
    int cnt = jhi - jlo + 1;
    float s = -1e30f;
    if (lane < cnt) {
      const v8h* qp = (const v8h*)(sQ + i * LDX + h * CD_HD);
      const v4f* kp = (const v4f*)(sSend + (h * 16 + jlo + lane) * CD_HD);
      float a = 0.f;
#pragma unroll
      for (int d8 = 0; d8 < CD_HD / 8; ++d8) {
        v8h qv = qp[d8];
        v4f k0 = kp[2 * d8], k1 = kp[2 * d8 + 1];
#pragma unroll
        for (int u = 0; u < 4; ++u) a += (float)qv[u] * k0[u];
#pragma unroll
        for (int u = 0; u < 4; ++u) a += (float)qv[4 + u] * k1[u];
      }
      a *= CD_SCALE;
      s = fminf(fmaxf(a, -10000.f), 10000.f);
    }
    float m = wave_max(s);
    float e = (lane < cnt) ? __expf(s - m) : 0.f;
    float den = wave_sum(e);
    float w = e / den;
#pragma unroll
    for (int q = 0; q < 4; ++q) {
      int d = q * 32 + lane;
      float o = 0.f;
      for (int jj = 0; jj < cnt; ++jj)
        o += __shfl(w, jj, 32) * sSend[(h * 16 + jlo + jj) * CD_HD + d];
      sRecv[i * LDX + h * CD_HD + d] = (h16)o;
    }
  }
  __syncthreads();

  // ---- phase 7: MLP2(recv) (overlaps dead K/V/Q regions) --------------
  mlp_block(sRecv, m2w1, m2b1, m2w2, m2b2, sH, sAcc, wv, lane);
  __syncthreads();

  // ---- phase 8: out = LN3(LN2(x_proj + MLP2(recv))) -------------------
  for (int r = wv; r < CD_T; r += 8) {
    float vals[8], s0 = 0.f, s1 = 0.f;
#pragma unroll
    for (int q = 0; q < 8; ++q) {
      int d = lane + q * 32;
      float v = (float)sXp[r * LDX + d] + sAcc[r * LDACC + d];
      vals[q] = v; s0 += v; s1 += v * v;
    }
    s0 = wave_sum(s0); s1 = wave_sum(s1);
    float mean = s0 * (1.0f / 256.0f);
    float inv  = rsqrtf(s1 * (1.0f / 256.0f) - mean * mean + CD_EPS);
    float t0 = 0.f, t1 = 0.f;
#pragma unroll
    for (int q = 0; q < 8; ++q) {
      int d = lane + q * 32;
      float y = (vals[q] - mean) * inv * g2[d] + be2[d];
      vals[q] = y; t0 += y; t1 += y * y;
    }
    t0 = wave_sum(t0); t1 = wave_sum(t1);
    float mean2 = t0 * (1.0f / 256.0f);
    float inv2  = rsqrtf(t1 * (1.0f / 256.0f) - mean2 * mean2 + CD_EPS);
    float* og = out + ((size_t)(b * CD_T + r) * CD_L + l) * CD_D;
#pragma unroll
    for (int q = 0; q < 8; ++q) {
      int d = lane + q * 32;
      og[d] = (vals[q] - mean2) * inv2 * g3[d] + be3[d];
    }
  }
}

// ---- weight conversion helpers --------------------------------------------
__global__ void cvt16(const float* __restrict__ s, h16* __restrict__ d, int n) {
  int i = blockIdx.x * blockDim.x + threadIdx.x;
  if (i < n) d[i] = (h16)s[i];
}

// router [L][10][256] f32 -> padded [L][16][256] f16 (rows 10..15 zero)
__global__ void cvt_router(const float* __restrict__ s, h16* __restrict__ d) {
  int i = blockIdx.x * blockDim.x + threadIdx.x;   // L*16*256 total
  int dcol = i & 255;
  int r    = (i >> 8) & 15;
  int l    = i >> 12;
  d[i] = (r < CD_F) ? (h16)s[((size_t)l * CD_F + r) * 256 + dcol] : (h16)0.0f;
}

// ---------------------------------------------------------------------------
extern "C" void kernel_launch(void* const* d_in, const int* in_sizes, int n_in,
                              void* d_out, int out_size, void* d_ws, size_t ws_size,
                              hipStream_t stream) {
  (void)in_sizes; (void)n_in; (void)out_size; (void)ws_size;
  const float* x      = (const float*)d_in[0];
  const float* router = (const float*)d_in[1];
  const float* Wr = (const float*)d_in[2];  const float* br = (const float*)d_in[3];
  const float* Wk = (const float*)d_in[4];  const float* bk = (const float*)d_in[5];
  const float* Wv = (const float*)d_in[6];  const float* bv = (const float*)d_in[7];
  const float* Wq = (const float*)d_in[8];  const float* bq = (const float*)d_in[9];
  const float* g1 = (const float*)d_in[10]; const float* be1 = (const float*)d_in[11];
  const float* g2 = (const float*)d_in[12]; const float* be2 = (const float*)d_in[13];
  const float* g3 = (const float*)d_in[14]; const float* be3 = (const float*)d_in[15];
  const float* m1w1 = (const float*)d_in[16]; const float* m1b1 = (const float*)d_in[17];
  const float* m1w2 = (const float*)d_in[18]; const float* m1b2 = (const float*)d_in[19];
  const float* m2w1 = (const float*)d_in[20]; const float* m2b1 = (const float*)d_in[21];
  const float* m2w2 = (const float*)d_in[22]; const float* m2b2 = (const float*)d_in[23];
  float* outp = (float*)d_out;

  // f16 weight arena in d_ws
  h16* ws = (h16*)d_ws;
  size_t o = 0;
  h16* wr16 = ws + o; o += 256 * 256;
  h16* wk16 = ws + o; o += 256 * 256;
  h16* wv16 = ws + o; o += 256 * 256;
  h16* wq16 = ws + o; o += 256 * 256;
  h16* w11  = ws + o; o += CD_DFF * 256;   // m1_w1 [DFF][D]
  h16* w12  = ws + o; o += 256 * CD_DFF;   // m1_w2 [D][DFF]
  h16* w21  = ws + o; o += CD_DFF * 256;
  h16* w22  = ws + o; o += 256 * CD_DFF;
  h16* rt16 = ws + o; o += (size_t)CD_L * 16 * 256;

  const int nd2  = 256 * 256;
  const int ndff = 256 * CD_DFF;
  cvt16<<<(nd2 + 255) / 256, 256, 0, stream>>>(Wr, wr16, nd2);
  cvt16<<<(nd2 + 255) / 256, 256, 0, stream>>>(Wk, wk16, nd2);
  cvt16<<<(nd2 + 255) / 256, 256, 0, stream>>>(Wv, wv16, nd2);
  cvt16<<<(nd2 + 255) / 256, 256, 0, stream>>>(Wq, wq16, nd2);
  cvt16<<<(ndff + 255) / 256, 256, 0, stream>>>(m1w1, w11, ndff);
  cvt16<<<(ndff + 255) / 256, 256, 0, stream>>>(m1w2, w12, ndff);
  cvt16<<<(ndff + 255) / 256, 256, 0, stream>>>(m2w1, w21, ndff);
  cvt16<<<(ndff + 255) / 256, 256, 0, stream>>>(m2w2, w22, ndff);
  cvt_router<<<(CD_L * 16 * 256) / 256, 256, 0, stream>>>(router, rt16);

  (void)hipFuncSetAttribute(reinterpret_cast<const void*>(cda_main),
                            hipFuncAttributeMaxDynamicSharedMemorySize, SMEM_BYTES);
  cda_main<<<dim3(CD_B * CD_L), dim3(256), SMEM_BYTES, stream>>>(
      x, rt16, wr16, br, wk16, bk, wv16, bv, wq16, bq,
      g1, be1, g2, be2, g3, be3,
      w11, m1b1, w12, m1b2, w21, m2b1, w22, m2b2, outp);
}
